// EETQLinear_7000796692999
// MI455X (gfx1250) — matrix operand order repaired and checked by reading, not compiled, not run on hardware
//
#include <hip/hip_runtime.h>
#include <stdint.h>

// ---- CDNA5 WMMA operand types (wave32) ----
typedef __attribute__((ext_vector_type(16))) _Float16 v16h;
typedef __attribute__((ext_vector_type(8)))  float    v8f;
typedef __attribute__((ext_vector_type(8)))  _Float16 v8h;

// ---- problem dims (reference: B=4, S=2048, IN=4096, OUT=4096) ----
constexpr int M_DIM = 8192;
constexpr int N_DIM = 4096;
constexpr int K_DIM = 4096;

// ---- GEMM tiling: 128x128 block tile, K-tile 64, 8 waves (4x2), 32x64 per wave ----
constexpr int BM = 128, BN = 128, BK = 64;

union FragU { v16h v; uint4 u[2]; };

// ============================================================================
// Pre-pass: Wt[n,k] = fp16(Wq[k,n]) * scale[n]   (one-time dequant + transpose)
// 64x64 int8 tiles through LDS; output rows are k-contiguous so the GEMM can
// async-copy B fragments directly.
// ============================================================================
__global__ __launch_bounds__(256)
void dequant_transpose_kernel(const int8_t*   __restrict__ Wq,    // [K, N] int8
                              const _Float16* __restrict__ scale, // [N]
                              _Float16*       __restrict__ Wt)    // [N, K] fp16
{
    __shared__ int8_t lds[64][80];            // pad to 80 keeps 16B-aligned rows
    const int kT = blockIdx.y * 64;
    const int nT = blockIdx.x * 64;
    const int t  = threadIdx.x;

    {   // load 64x64 int8 tile, coalesced along n
        const int r = t >> 2, c = (t & 3) * 16;
        uint4 d = *reinterpret_cast<const uint4*>(Wq + (size_t)(kT + r) * N_DIM + nT + c);
        *reinterpret_cast<uint4*>(&lds[r][c]) = d;
    }
    __syncthreads();

    #pragma unroll
    for (int it = 0; it < 2; ++it) {          // write 64x64 fp16, coalesced along k
        const int idx = t + it * 256;         // 0..511
        const int n   = idx >> 3;
        const int c8  = (idx & 7) * 8;        // k offset
        const _Float16 s = scale[nT + n];
        v8h o;
        #pragma unroll
        for (int i = 0; i < 8; ++i)
            o[i] = (_Float16)((_Float16)lds[c8 + i][n] * s);   // fp16 mul, as reference
        *reinterpret_cast<v8h*>(Wt + (size_t)(nT + n) * K_DIM + kT + c8) = o;
    }
}

// ============================================================================
// Main GEMM: fp16 WMMA, both tiles staged with global_load_async_to_lds_b128
// into packed WMMA-fragment order (512B chunks, lane l owns bytes [16l,16l+16)).
// ============================================================================
__global__ __launch_bounds__(256)
void w8a16_gemm_kernel(const _Float16* __restrict__ X,     // [M, K] fp16
                       const _Float16* __restrict__ Wt,    // [N, K] fp16 (pre-dequantized)
                       const _Float16* __restrict__ bias,  // [N] fp16
                       _Float16*       __restrict__ Out)   // [M, N] fp16
{
    __shared__ uint4 As[2][1024];   // 2 x 16 KB
    __shared__ uint4 Bs[2][1024];   // 2 x 16 KB

    const int tid  = threadIdx.x;
    const int lane = tid & 31;
    const int wave = tid >> 5;
    const int wm   = wave & 3;       // 32-row slice
    const int wn   = wave >> 2;      // 64-col slice
    const int l15  = lane & 15;
    const int hi   = lane >> 4;

    const int m0 = blockIdx.y * BM;
    const int n0 = blockIdx.x * BN;

    v8f acc[2][4];
    #pragma unroll
    for (int i = 0; i < 2; ++i)
        #pragma unroll
        for (int j = 0; j < 4; ++j)
            acc[i][j] = {};

    // A chunk (sm,kkI,c): lane l -> m = sm*16 + (l&15), k = kkI*32 + c*16 + hi*8 + {0..7}
    auto loadA_async = [&](int k0, int buf) {
        #pragma unroll
        for (int i = 0; i < 4; ++i) {
            const int q   = wave * 4 + i;
            const int sm  = q >> 2;
            const int kkI = (q >> 1) & 1;
            const int c   = q & 1;
            const uint64_t g = (uint64_t)(X + (size_t)(m0 + sm * 16 + l15) * K_DIM
                                            + k0 + kkI * 32 + c * 16 + hi * 8);
            const uint32_t l_off = (uint32_t)(uintptr_t)(&As[buf][q * 32 + lane]);
            asm volatile("global_load_async_to_lds_b128 %0, %1, off"
                         :: "v"(l_off), "v"(g) : "memory");
        }
    };

    // B chunk (sn,kkI,c): lane l -> n = sn*16 + (l&15), k = kkI*32 + c*16 + hi*8 + {0..7}
    auto loadB_async = [&](int k0, int buf) {
        #pragma unroll
        for (int i = 0; i < 4; ++i) {
            const int q   = wave * 4 + i;
            const int sn  = q >> 2;
            const int kkI = (q >> 1) & 1;
            const int c   = q & 1;
            const uint64_t g = (uint64_t)(Wt + (size_t)(n0 + sn * 16 + l15) * K_DIM
                                             + k0 + kkI * 32 + c * 16 + hi * 8);
            const uint32_t l_off = (uint32_t)(uintptr_t)(&Bs[buf][q * 32 + lane]);
            asm volatile("global_load_async_to_lds_b128 %0, %1, off"
                         :: "v"(l_off), "v"(g) : "memory");
        }
    };

    // 16 WMMAs per K-tile; fragments are conflict-free ds_load_b128 at lane*16
    auto compute = [&](int buf) {
        #pragma unroll
        for (int kkI = 0; kkI < 2; ++kkI) {
            FragU a[2], b[4];
            #pragma unroll
            for (int i = 0; i < 2; ++i) {
                const uint4* p = &As[buf][(((wm * 2 + i) * 2 + kkI) * 2) * 32 + lane];
                a[i].u[0] = p[0];
                a[i].u[1] = p[32];
            }
            #pragma unroll
            for (int j = 0; j < 4; ++j) {
                const uint4* p = &Bs[buf][(((wn * 4 + j) * 2 + kkI) * 2) * 32 + lane];
                b[j].u[0] = p[0];
                b[j].u[1] = p[32];
            }
            #pragma unroll
            for (int i = 0; i < 2; ++i)
                #pragma unroll
                for (int j = 0; j < 4; ++j)
                    acc[i][j] = __builtin_amdgcn_wmma_f32_16x16x32_f16(
                        false, a[i].v, false, b[j].v,
                        (short)0, acc[i][j], false, false);
        }
    };

    // ---- prologue ----
    loadA_async(0, 0);
    loadB_async(0, 0);
    asm volatile("s_wait_asynccnt 0x0" ::: "memory");
    __syncthreads();

    // ---- main loop: tile k+1's async copies run under tile k's WMMAs ----
    for (int k0 = 0; k0 < K_DIM; k0 += BK) {
        const int  buf  = (k0 >> 6) & 1;
        const bool more = (k0 + BK) < K_DIM;
        if (more) {
            loadA_async(k0 + BK, buf ^ 1);
            loadB_async(k0 + BK, buf ^ 1);
        }
        compute(buf);
        if (more)
            asm volatile("s_wait_asynccnt 0x0" ::: "memory");
        __syncthreads();
    }

    // ---- epilogue: scale already folded into Wt, just add bias ----
    #pragma unroll
    for (int j = 0; j < 4; ++j) {
        const int n    = n0 + wn * 64 + j * 16 + l15;
        const float bz = (float)bias[n];
        #pragma unroll
        for (int i = 0; i < 2; ++i) {
            const int mbase = m0 + wm * 32 + i * 16 + hi * 8;
            #pragma unroll
            for (int r = 0; r < 8; ++r) {
                float v = acc[i][j][r];
                Out[(size_t)(mbase + r) * N_DIM + n] = (_Float16)(v + bz);
            }
        }
    }
}

extern "C" void kernel_launch(void* const* d_in, const int* in_sizes, int n_in,
                              void* d_out, int out_size, void* d_ws, size_t ws_size,
                              hipStream_t stream) {
    const _Float16* x   = (const _Float16*)d_in[0];   // fp16 [4,2048,4096]
    const int8_t*   wq  = (const int8_t*)  d_in[1];   // int8 [4096,4096]
    const _Float16* sc  = (const _Float16*)d_in[2];   // fp16 [4096]
    const _Float16* bi  = (const _Float16*)d_in[3];   // fp16 [4096]
    _Float16*       out = (_Float16*)d_out;           // fp16 [4,2048,4096]
    _Float16*       wt  = (_Float16*)d_ws;            // fp16 [4096,4096] = 32 MiB scratch

    // 1) one-time dequant + transpose (+scale fold): Wq[k,n] -> Wt[n,k]
    dim3 tgrid(N_DIM / 64, K_DIM / 64);               // 64 x 64 tiles
    dequant_transpose_kernel<<<tgrid, dim3(256), 0, stream>>>(wq, sc, wt);

    // 2) compute-bound fp16 WMMA GEMM + bias
    dim3 grid(N_DIM / BN, M_DIM / BM);                // 32 x 64 workgroups
    w8a16_gemm_kernel<<<grid, dim3(256), 0, stream>>>(x, wt, bi, out);
}
